// VN_DGCNN_59021440582193
// MI455X (gfx1250) — matrix operand order repaired and checked by reading, not compiled, not run on hardware
//
#include <hip/hip_runtime.h>

#define EMBED   64
#define KNN     20
#define NLAYERS 4
#define EPSV    1e-8f
#define BATCH   4
#define NPTS    4096

typedef __attribute__((ext_vector_type(16))) _Float16 v16h;
typedef __attribute__((ext_vector_type(8)))  _Float16 v8h;
typedef __attribute__((ext_vector_type(8)))  float    v8f;

// ---------------------------------------------------------------------------
// Weight prep: fp32 -> f16, transposed to column-major so each WMMA B-fragment
// (K = hi*16 + h contiguous, N = lane&15) is one contiguous 32-byte load.
// w1t layout: [layer][col(64)][k(128)]   w2t: [layer][col(64)][k(64)]
// ---------------------------------------------------------------------------
__global__ void vn_wconv_kernel(const float* __restrict__ wa,
                                const float* __restrict__ wb,
                                _Float16* __restrict__ w1t,
                                _Float16* __restrict__ w2t) {
  int t = blockIdx.x * blockDim.x + threadIdx.x;
  const int n1 = NLAYERS * 64 * 128;
  if (t < n1) {
    int layer = t / (64 * 128);
    int r     = t % (64 * 128);
    int col   = r / 128;
    int k     = r % 128;
    w1t[t] = (_Float16)wa[layer * 128 * 64 + k * 64 + col];
    return;
  }
  int t2 = t - n1;
  if (t2 < NLAYERS * 64 * 64) {
    int layer = t2 / (64 * 64);
    int r     = t2 % (64 * 64);
    int col   = r / 64;
    int k     = r % 64;
    w2t[t2] = (_Float16)wb[layer * 64 * 64 + k * 64 + col];
  }
}

// ---------------------------------------------------------------------------
// Initial embed + vn_relu:  h[b,n,v,e] = x[b,n,v]*w0[e] * norm/(norm+eps)
// where norm = |x[b,n]| * |w0[e]| (vn_relu of a rank-1 lift).
// ---------------------------------------------------------------------------
__global__ void vn_embed_kernel(const float* __restrict__ x,
                                const float* __restrict__ w0,
                                float* __restrict__ h) {
  int t = blockIdx.x * blockDim.x + threadIdx.x;   // over B*N*EMBED
  if (t >= BATCH * NPTS * EMBED) return;
  int e  = t % EMBED;
  int bn = t / EMBED;
  float x0 = x[bn * 3 + 0], x1 = x[bn * 3 + 1], x2 = x[bn * 3 + 2];
  float w  = w0[e];
  float nx = sqrtf(x0 * x0 + x1 * x1 + x2 * x2);
  float nm = nx * fabsf(w);
  float s  = nm / (nm + EPSV);
  float*  o = h + (size_t)bn * 192;
  o[e]        = x0 * w * s;
  o[64 + e]   = x1 * w * s;
  o[128 + e]  = x2 * w * s;
}

// ---------------------------------------------------------------------------
// positions[b,n,v] = mean_e h[b,n,v,e]
// ---------------------------------------------------------------------------
__global__ void vn_pos_kernel(const float* __restrict__ h,
                              float* __restrict__ pos) {
  int t = blockIdx.x * blockDim.x + threadIdx.x;   // over B*N*3
  if (t >= BATCH * NPTS * 3) return;
  const float* row = h + (size_t)t * 64;
  float s = 0.f;
  #pragma unroll 8
  for (int e = 0; e < 64; ++e) s += row[e];
  pos[t] = s * (1.0f / 64.0f);
}

// ---------------------------------------------------------------------------
// KNN (k=20, squared L2, self excluded). One batch's positions live in LDS.
// Top-20 kept in registers with a fully-unrolled branch-free sorted insert
// (static indices only -> stays in VGPRs, no scratch).
// ---------------------------------------------------------------------------
__global__ __launch_bounds__(256) void vn_knn_kernel(const float* __restrict__ pos,
                                                     int* __restrict__ idx) {
  __shared__ float sp[NPTS * 3];
  const int b     = blockIdx.x / (NPTS / 256);
  const int chunk = blockIdx.x % (NPTS / 256);
  const float* pb = pos + (size_t)b * NPTS * 3;
  for (int i = threadIdx.x; i < NPTS * 3; i += 256) sp[i] = pb[i];
  __syncthreads();

  const int i = chunk * 256 + threadIdx.x;
  const float px = sp[i * 3 + 0], py = sp[i * 3 + 1], pz = sp[i * 3 + 2];

  float bd[KNN];
  int   bi[KNN];
  #pragma unroll
  for (int k = 0; k < KNN; ++k) { bd[k] = 3.4e38f; bi[k] = 0; }

  for (int j = 0; j < NPTS; ++j) {
    float dx = px - sp[j * 3 + 0];
    float dy = py - sp[j * 3 + 1];
    float dz = pz - sp[j * 3 + 2];
    float d  = dx * dx + dy * dy + dz * dz;
    if (j == i) continue;
    if (d < bd[KNN - 1]) {
      // branch-free sorted insert, descending pass over static slots.
      #pragma unroll
      for (int k = KNN - 1; k >= 1; --k) {
        float pk = bd[k - 1]; int pik = bi[k - 1];
        if (bd[k] > d) {
          bool prev_le = (pk <= d);
          bd[k] = prev_le ? d : pk;
          bi[k] = prev_le ? j : pik;
        }
      }
      if (bd[0] > d) { bd[0] = d; bi[0] = j; }
    }
  }
  int* o = idx + ((size_t)b * NPTS + i) * KNN;
  #pragma unroll
  for (int k = 0; k < KNN; ++k) o[k] = bi[k];
}

// ---------------------------------------------------------------------------
// Edge conv: one node per 64-thread (2-wave) workgroup.
//   E (64x128 f16, rows 60..63 zero) @ W1 (128x64) -> vn_relu ->
//   @ W2 (64x64) -> max-|.| over K -> residual + vn_layernorm -> hout
// GEMMs via v_wmma_f32_16x16x32_f16; each wave owns 2 M-tiles x 4 N-tiles.
// ---------------------------------------------------------------------------
__global__ __launch_bounds__(64) void vn_edgeconv_kernel(
    const float* __restrict__ h, const int* __restrict__ idx,
    const _Float16* __restrict__ w1t, const _Float16* __restrict__ w2t,
    float* __restrict__ hout, int layer) {
  __shared__ __align__(32) float    hc[192];
  __shared__              int      nidx[KNN];
  __shared__ __align__(32) _Float16 E[64 * 128];
  __shared__ __align__(32) float    G[64 * 64];
  __shared__ __align__(32) _Float16 E2[64 * 64];
  __shared__              float    nbuf[EMBED];

  const int tid  = threadIdx.x;
  const int node = blockIdx.x;                 // b*N + n
  const int b    = node >> 12;                 // N = 4096
  const float* hb = h + (size_t)b * NPTS * 192;

  // ---- stage 0: center features + neighbor indices ----
  for (int i = tid; i < 192; i += 64) hc[i] = h[(size_t)node * 192 + i];
  if (tid < KNN) nidx[tid] = idx[(size_t)node * KNN + tid];
  __syncthreads();

  // ---- stage 1: build edge matrix E (rows 0..59 = edge k, vec comp v) ----
  for (int i = tid; i < KNN * 3 * 64; i += 64) {
    int e   = i & 63;
    int row = i >> 6;
    int k   = row / 3;
    int v   = row - 3 * k;
    float c  = hc[v * 64 + e];
    float nb = hb[(size_t)nidx[k] * 192 + v * 64 + e];
    E[row * 128 + e]      = (_Float16)c;
    E[row * 128 + 64 + e] = (_Float16)(nb - c);
  }
  for (int i = tid; i < 4 * 128; i += 64) E[60 * 128 + i] = (_Float16)0.f;
  __syncthreads();

  const int wave = tid >> 5;
  const int lane = tid & 31;
  const int hi   = (lane >> 4) & 1;
  const int lcol = lane & 15;

  // ---- stage 2: GEMM1  G = E(64x128) @ W1(128x64) ----
  v8f acc[2][4];
  #pragma unroll
  for (int mt = 0; mt < 2; ++mt)
    #pragma unroll
    for (int nt = 0; nt < 4; ++nt) acc[mt][nt] = {};

  const _Float16* W1 = w1t + layer * 64 * 128;   // [col][k]
  #pragma unroll
  for (int kk = 0; kk < 4; ++kk) {
    const int k0 = kk * 32;
    v16h afrag[2];
    #pragma unroll
    for (int mt = 0; mt < 2; ++mt) {
      const int row = wave * 32 + mt * 16 + lcol;
      const _Float16* base = &E[row * 128 + k0 + hi * 16];
      v8h lo = *(const v8h*)(base);        // K = k0 + hi*16 .. +7
      v8h hh = *(const v8h*)(base + 8);    // K = k0 + hi*16 + 8 .. +15
      afrag[mt] = __builtin_shufflevector(lo, hh, 0, 1, 2, 3, 4, 5, 6, 7,
                                          8, 9, 10, 11, 12, 13, 14, 15);
    }
    #pragma unroll
    for (int nt = 0; nt < 4; ++nt) {
      const int col = nt * 16 + lcol;
      v16h bfrag = *(const v16h*)(W1 + col * 128 + k0 + hi * 16);
      #pragma unroll
      for (int mt = 0; mt < 2; ++mt)
        acc[mt][nt] = __builtin_amdgcn_wmma_f32_16x16x32_f16(
            false, afrag[mt], false, bfrag, (short)0, acc[mt][nt], false, false);
    }
  }

  // accumulators -> LDS (f32 64x64 row-major)
  #pragma unroll
  for (int mt = 0; mt < 2; ++mt)
    #pragma unroll
    for (int nt = 0; nt < 4; ++nt)
      #pragma unroll
      for (int r = 0; r < 8; ++r) {
        const int row = wave * 32 + mt * 16 + hi * 8 + r;
        G[row * 64 + nt * 16 + lcol] = acc[mt][nt][r];
      }
  __syncthreads();

  // ---- stage 3: vn_relu over each edge's 3-row triple -> E2 (f16) ----
  for (int i = tid; i < KNN * 64; i += 64) {
    int e = i & 63;
    int k = i >> 6;
    float g0 = G[(3 * k + 0) * 64 + e];
    float g1 = G[(3 * k + 1) * 64 + e];
    float g2 = G[(3 * k + 2) * 64 + e];
    float n  = sqrtf(g0 * g0 + g1 * g1 + g2 * g2);
    float s  = n / (n + EPSV);
    E2[(3 * k + 0) * 64 + e] = (_Float16)(g0 * s);
    E2[(3 * k + 1) * 64 + e] = (_Float16)(g1 * s);
    E2[(3 * k + 2) * 64 + e] = (_Float16)(g2 * s);
  }
  for (int i = tid; i < 4 * 64; i += 64) E2[60 * 64 + i] = (_Float16)0.f;
  __syncthreads();

  // ---- stage 4: GEMM2  G = E2(64x64) @ W2(64x64) ----
  v8f acc2[2][4];
  #pragma unroll
  for (int mt = 0; mt < 2; ++mt)
    #pragma unroll
    for (int nt = 0; nt < 4; ++nt) acc2[mt][nt] = {};

  const _Float16* W2 = w2t + layer * 64 * 64;    // [col][k]
  #pragma unroll
  for (int kk = 0; kk < 2; ++kk) {
    const int k0 = kk * 32;
    v16h afrag[2];
    #pragma unroll
    for (int mt = 0; mt < 2; ++mt) {
      const int row = wave * 32 + mt * 16 + lcol;
      const _Float16* base = &E2[row * 64 + k0 + hi * 16];
      v8h lo = *(const v8h*)(base);
      v8h hh = *(const v8h*)(base + 8);
      afrag[mt] = __builtin_shufflevector(lo, hh, 0, 1, 2, 3, 4, 5, 6, 7,
                                          8, 9, 10, 11, 12, 13, 14, 15);
    }
    #pragma unroll
    for (int nt = 0; nt < 4; ++nt) {
      const int col = nt * 16 + lcol;
      v16h bfrag = *(const v16h*)(W2 + col * 64 + k0 + hi * 16);
      #pragma unroll
      for (int mt = 0; mt < 2; ++mt)
        acc2[mt][nt] = __builtin_amdgcn_wmma_f32_16x16x32_f16(
            false, afrag[mt], false, bfrag, (short)0, acc2[mt][nt], false, false);
    }
  }

  #pragma unroll
  for (int mt = 0; mt < 2; ++mt)
    #pragma unroll
    for (int nt = 0; nt < 4; ++nt)
      #pragma unroll
      for (int r = 0; r < 8; ++r) {
        const int row = wave * 32 + mt * 16 + hi * 8 + r;
        G[row * 64 + nt * 16 + lcol] = acc2[mt][nt][r];
      }
  __syncthreads();

  // ---- stage 5: per-channel argmax over K, residual, vn_layernorm ----
  {
    const int e = tid;                   // one channel per thread
    float best = -1.0f;
    int   bk   = 0;
    #pragma unroll
    for (int k = 0; k < KNN; ++k) {
      float g0 = G[(3 * k + 0) * 64 + e];
      float g1 = G[(3 * k + 1) * 64 + e];
      float g2 = G[(3 * k + 2) * 64 + e];
      float sq = g0 * g0 + g1 * g1 + g2 * g2;
      if (sq > best) { best = sq; bk = k; }   // first max wins (matches argmax)
    }
    float h0v = hc[0 * 64 + e] + G[(3 * bk + 0) * 64 + e];
    float h1v = hc[1 * 64 + e] + G[(3 * bk + 1) * 64 + e];
    float h2v = hc[2 * 64 + e] + G[(3 * bk + 2) * 64 + e];
    float nrm = sqrtf(h0v * h0v + h1v * h1v + h2v * h2v);
    nbuf[e] = nrm;
    __syncthreads();
    float s1 = 0.f, s2 = 0.f;
    #pragma unroll 8
    for (int q = 0; q < 64; ++q) { float v = nbuf[q]; s1 += v; s2 += v * v; }
    float mean = s1 * (1.0f / 64.0f);
    float var  = s2 * (1.0f / 64.0f) - mean * mean;
    var = var < 0.f ? 0.f : var;
    float stdv  = sqrtf(var) + EPSV;
    float scale = (nrm - mean) / stdv / (nrm + EPSV);
    float* o = hout + (size_t)node * 192;
    o[e]       = h0v * scale;
    o[64 + e]  = h1v * scale;
    o[128 + e] = h2v * scale;
  }
}

// ---------------------------------------------------------------------------
extern "C" void kernel_launch(void* const* d_in, const int* in_sizes, int n_in,
                              void* d_out, int out_size, void* d_ws, size_t ws_size,
                              hipStream_t stream) {
  const float* x  = (const float*)d_in[0];
  const float* w0 = (const float*)d_in[1];
  const float* wa = (const float*)d_in[2];
  const float* wb = (const float*)d_in[3];
  float* out = (float*)d_out;

  char* ws = (char*)d_ws;
  size_t off = 0;
  float* h_ws = (float*)(ws + off); off += (size_t)BATCH * NPTS * 192 * sizeof(float);
  float* pos  = (float*)(ws + off); off += (size_t)BATCH * NPTS * 3 * sizeof(float);
  int*   idx  = (int*)(ws + off);   off += (size_t)BATCH * NPTS * KNN * sizeof(int);
  off = (off + 255) & ~(size_t)255;
  _Float16* w1t = (_Float16*)(ws + off); off += (size_t)NLAYERS * 64 * 128 * sizeof(_Float16);
  _Float16* w2t = (_Float16*)(ws + off); off += (size_t)NLAYERS * 64 * 64 * sizeof(_Float16);

  const int nw = NLAYERS * (64 * 128 + 64 * 64);
  vn_wconv_kernel<<<(nw + 255) / 256, 256, 0, stream>>>(wa, wb, w1t, w2t);
  vn_embed_kernel<<<(BATCH * NPTS * EMBED + 255) / 256, 256, 0, stream>>>(x, w0, out);

  // ping-pong: embed->d_out, L1->ws, L2->d_out, L3->ws, L4->d_out (final)
  float* cur = out;
  float* nxt = h_ws;
  for (int layer = 0; layer < NLAYERS; ++layer) {
    vn_pos_kernel<<<(BATCH * NPTS * 3 + 255) / 256, 256, 0, stream>>>(cur, pos);
    vn_knn_kernel<<<BATCH * (NPTS / 256), 256, 0, stream>>>(pos, idx);
    vn_edgeconv_kernel<<<BATCH * NPTS, 64, 0, stream>>>(cur, idx, w1t, w2t, nxt, layer);
    float* t = cur; cur = nxt; nxt = t;
  }
}
